// FDB_65979287601425
// MI455X (gfx1250) — matrix operand order repaired and checked by reading, not compiled, not run on hardware
//
#include <hip/hip_runtime.h>
#include <stdint.h>

typedef __attribute__((ext_vector_type(16))) __bf16 v16bf;
typedef __attribute__((ext_vector_type(8)))  float  v8f;

#define BATCH 16
#define C_DIM 64
#define WIDTH 210
#define HW    44100
#define KREAL 84
#define KP    96
#define TILE  128
#define NWAVE 8

union AFrag { v16bf v; __bf16 h[16]; uint32_t u[8]; };

// ---- monotonic float<->uint key for atomic max (handles negatives) ----
__device__ __forceinline__ uint32_t fkey(float f) {
    uint32_t u = __float_as_uint(f);
    return (u & 0x80000000u) ? ~u : (u | 0x80000000u);
}
__device__ __forceinline__ float funkey(uint32_t k) {
    uint32_t u = (k & 0x80000000u) ? (k & 0x7fffffffu) : ~k;
    return __uint_as_float(u);
}

// =======================================================================
// Kernel 1: pyramid max-pool. One block per (b, tensor, channel).
// All 4 grids (7,5,3,1) in a single pass; LDS atomic-max on uint keys.
// =======================================================================
__global__ __launch_bounds__(256) void psp_pool_kernel(
    const float* __restrict__ a, const float* __restrict__ c,
    float* __restrict__ pout)
{
    __shared__ uint32_t bins[84];
    int bidx = blockIdx.x;
    int ch = bidx & 63;
    int t  = (bidx >> 6) & 1;
    int b  = bidx >> 7;
    const float* src = (t == 0 ? a : c) + ((size_t)b * C_DIM + ch) * HW;
    if (threadIdx.x < 84) bins[threadIdx.x] = 0u;
    __syncthreads();
    for (int idx = threadIdx.x; idx < HW; idx += 256) {
        int h = idx / WIDTH;
        int w = idx - h * WIDTH;
        uint32_t k = fkey(src[idx]);
        atomicMax(&bins[(h / 30) * 7 + (w / 30)], k);
        atomicMax(&bins[49 + (h / 42) * 5 + (w / 42)], k);
        atomicMax(&bins[74 + (h / 70) * 3 + (w / 70)], k);
        atomicMax(&bins[83], k);
    }
    __syncthreads();
    if (threadIdx.x < 84)
        pout[((size_t)(b * 2 + t) * C_DIM + ch) * KREAL + threadIdx.x] =
            funkey(bins[threadIdx.x]);
}

// =======================================================================
// Kernel 2: per (b,branch) fold weights into Kq^T (bf16 [k][c]),
// M^T (bf16 [o][k], output bias folded in), d (f32, -1e30 k-padding).
// =======================================================================
__global__ __launch_bounds__(256) void prep_kernel(
    int br, const float* __restrict__ pool,
    const float* __restrict__ wq, const float* __restrict__ bq,
    const float* __restrict__ wk, const float* __restrict__ bk,
    const float* __restrict__ wv, const float* __restrict__ bv,
    const float* __restrict__ wo, const float* __restrict__ bo,
    __bf16* __restrict__ Kqt, float* __restrict__ dvec,
    __bf16* __restrict__ Mt)
{
    __shared__ float keyS[C_DIM * KREAL];
    __shared__ float valS[C_DIM * KREAL];
    int b = blockIdx.x;
    int img = b * 2 + br;
    const float* p = pool + (size_t)img * C_DIM * KREAL;

    for (int idx = threadIdx.x; idx < C_DIM * KREAL; idx += 256) {
        int o = idx / KREAL, k = idx - o * KREAL;
        float sk = bk[o], sv = bv[o];
        for (int cc = 0; cc < C_DIM; ++cc) {
            float pv = p[cc * KREAL + k];
            sk += wk[o * C_DIM + cc] * pv;
            sv += wv[o * C_DIM + cc] * pv;
        }
        keyS[idx] = sk;
        valS[idx] = sv;
    }
    __syncthreads();

    __bf16* Kq = Kqt + (size_t)img * KP * C_DIM;
    __bf16* M  = Mt  + (size_t)img * C_DIM * KP;
    float*  dd = dvec + (size_t)img * KP;

    // Kq^T[k][c] = sum_o wq[o][c] * key[o][k]   (zero pad k>=84)
    for (int idx = threadIdx.x; idx < KP * C_DIM; idx += 256) {
        int kk = idx / C_DIM, cc = idx - kk * C_DIM;
        float v = 0.f;
        if (kk < KREAL)
            for (int o = 0; o < C_DIM; ++o)
                v += wq[o * C_DIM + cc] * keyS[o * KREAL + kk];
        Kq[idx] = (__bf16)v;
    }
    // M^T[o][k] = sum_c wo[o][c]*val[c][k] + bo[o]  (zero pad k>=84)
    for (int idx = threadIdx.x; idx < C_DIM * KP; idx += 256) {
        int o = idx / KP, kk = idx - o * KP;
        float v = 0.f;
        if (kk < KREAL) {
            v = bo[o];
            for (int cc = 0; cc < C_DIM; ++cc)
                v += wo[o * C_DIM + cc] * valS[cc * KREAL + kk];
        }
        M[idx] = (__bf16)v;
    }
    // d[k] = sum_o bq[o]*key[o][k], -1e30 padding
    for (int kk = threadIdx.x; kk < KP; kk += 256) {
        float v = -1e30f;
        if (kk < KREAL) {
            v = 0.f;
            for (int o = 0; o < C_DIM; ++o)
                v += bq[o] * keyS[o * KREAL + kk];
        }
        dd[kk] = v;
    }
}

// =======================================================================
// Kernel 3: fused attention. 8 waves x 16 pixels = 128-pixel tile.
// logits = x^T Kq + d  (bf16 WMMA, f32 acc) -> softmax -> ctx^T = s M^T
// epilogue: out = x + a*(1+ctx_c) + c*(1+ctx_a), coalesced.
// =======================================================================
__global__ __launch_bounds__(256) void attn_kernel(
    const float* __restrict__ x, const float* __restrict__ a,
    const float* __restrict__ c,
    const __bf16* __restrict__ Kqt, const float* __restrict__ dvec,
    const __bf16* __restrict__ Mt, float* __restrict__ out)
{
    __shared__ __bf16 sS[NWAVE * 16 * KP];       // 24 KB: per-wave softmax rows
    __shared__ __bf16 ctxS[2 * C_DIM * TILE];    // 32 KB: ctx transposed stage

    const int tilesPerB = (HW + TILE - 1) / TILE;   // 345
    int b    = blockIdx.x / tilesPerB;
    int tile = blockIdx.x - b * tilesPerB;
    int base = tile * TILE;
    int wave = threadIdx.x >> 5;
    int lane = threadIdx.x & 31;
    int hl   = lane >> 4;          // wave half
    int ln   = lane & 15;
    int nb   = base + wave * 16;   // this wave's 16 pixels

    const float* xb = x + (size_t)b * C_DIM * HW;
    int nA = nb + ln;
    if (nA > HW - 1) nA = HW - 1;  // clamp tail (stores guarded later)

    // A-fragments of x^T in contiguous k order (B uses the same):
    // lane -> m = ln ; element e -> K = kc*32 + hl*16 + e
    AFrag ax[2];
#pragma unroll
    for (int kc = 0; kc < 2; ++kc)
#pragma unroll
        for (int e = 0; e < 16; ++e) {
            int chn = kc * 32 + hl * 16 + e;
            ax[kc].h[e] = (__bf16)xb[(size_t)chn * HW + nA];
        }

    __bf16* myS = sS + wave * 16 * KP;

#pragma unroll 1
    for (int br = 0; br < 2; ++br) {
        const __bf16* Kq = Kqt + (size_t)(b * 2 + br) * KP * C_DIM;
        const float*  dd = dvec + (size_t)(b * 2 + br) * KP;
        const __bf16* M  = Mt  + (size_t)(b * 2 + br) * C_DIM * KP;

        // ---- logits: 6 N-tiles of 16 k-columns, K=64 in two steps ----
        v8f acc[6];
#pragma unroll
        for (int nt = 0; nt < 6; ++nt) {
            v8f z = {0.f, 0.f, 0.f, 0.f, 0.f, 0.f, 0.f, 0.f};
            const v16bf* b0 = (const v16bf*)(Kq + (size_t)(nt * 16 + ln) * C_DIM + hl * 16);
            const v16bf* b1 = (const v16bf*)(Kq + (size_t)(nt * 16 + ln) * C_DIM + 32 + hl * 16);
            z = __builtin_amdgcn_wmma_f32_16x16x32_bf16(false, ax[0].v, false, *b0,
                                                        (short)0, z, false, false);
            z = __builtin_amdgcn_wmma_f32_16x16x32_bf16(false, ax[1].v, false, *b1,
                                                        (short)0, z, false, false);
            float dval = dd[nt * 16 + ln];   // same k for all 8 rows of this lane
#pragma unroll
            for (int r = 0; r < 8; ++r) z[r] += dval;
            acc[nt] = z;
        }

        // ---- softmax over k (rows m = r + 8*hl, 16 lanes per half) ----
        float mx[8], inv[8];
#pragma unroll
        for (int r = 0; r < 8; ++r) {
            float m0 = acc[0][r];
#pragma unroll
            for (int nt = 1; nt < 6; ++nt) m0 = fmaxf(m0, acc[nt][r]);
#pragma unroll
            for (int off = 1; off < 16; off <<= 1)
                m0 = fmaxf(m0, __shfl_xor(m0, off, 32));
            mx[r] = m0;
        }
#pragma unroll
        for (int r = 0; r < 8; ++r) {
            float s0 = 0.f;
#pragma unroll
            for (int nt = 0; nt < 6; ++nt) {
                float e0 = __expf(acc[nt][r] - mx[r]);
                acc[nt][r] = e0;
                s0 += e0;
            }
#pragma unroll
            for (int off = 1; off < 16; off <<= 1)
                s0 += __shfl_xor(s0, off, 32);
            inv[r] = 1.f / s0;
        }

        // ---- stage s through per-wave LDS (re-shape into A-fragments) ----
#pragma unroll
        for (int nt = 0; nt < 6; ++nt)
#pragma unroll
            for (int r = 0; r < 8; ++r)
                myS[(r + 8 * hl) * KP + nt * 16 + ln] = (__bf16)(acc[nt][r] * inv[r]);

        AFrag as[3];
#pragma unroll
        for (int kc = 0; kc < 3; ++kc)
#pragma unroll
            for (int e2 = 0; e2 < 8; ++e2) {
                int k = kc * 32 + hl * 16 + 2 * e2;
                as[kc].u[e2] = *(const uint32_t*)(myS + ln * KP + k);
            }

        // ---- ctx^T = s (16xKP) * M^T (KPx64): 4 o-tiles x 3 K-chunks ----
        __bf16* ctxBr = ctxS + br * C_DIM * TILE;
#pragma unroll
        for (int ot = 0; ot < 4; ++ot) {
            v8f z = {0.f, 0.f, 0.f, 0.f, 0.f, 0.f, 0.f, 0.f};
#pragma unroll
            for (int kc = 0; kc < 3; ++kc) {
                const v16bf* bm = (const v16bf*)(M + (size_t)(ot * 16 + ln) * KP + kc * 32 + hl * 16);
                z = __builtin_amdgcn_wmma_f32_16x16x32_bf16(false, as[kc].v, false, *bm,
                                                            (short)0, z, false, false);
            }
#pragma unroll
            for (int r = 0; r < 8; ++r)
                ctxBr[(ot * 16 + ln) * TILE + wave * 16 + r + 8 * hl] = (__bf16)z[r];
        }
    }

    __syncthreads();

    // ---- fused residual epilogue, coalesced over pixels ----
    const float* ab = a + (size_t)b * C_DIM * HW;
    const float* cb = c + (size_t)b * C_DIM * HW;
    float* ob = out + (size_t)b * C_DIM * HW;
    for (int idx = threadIdx.x; idx < C_DIM * TILE; idx += 256) {
        int chn = idx >> 7;          // / TILE
        int px  = idx & (TILE - 1);
        int n = base + px;
        if (n < HW) {
            size_t off = (size_t)chn * HW + n;
            float xv = xb[off], av = ab[off], cv = cb[off];
            float ca = (float)ctxS[(0 * C_DIM + chn) * TILE + px];
            float cc2 = (float)ctxS[(C_DIM + chn) * TILE + px];
            ob[off] = xv + av * (cc2 + 1.f) + cv * (ca + 1.f);
        }
    }
}

// =======================================================================
extern "C" void kernel_launch(void* const* d_in, const int* in_sizes, int n_in,
                              void* d_out, int out_size, void* d_ws, size_t ws_size,
                              hipStream_t stream)
{
    (void)in_sizes; (void)n_in; (void)out_size; (void)ws_size;
    const float* x   = (const float*)d_in[0];
    const float* a   = (const float*)d_in[1];
    const float* c   = (const float*)d_in[2];
    const float* wqa = (const float*)d_in[3];
    const float* bqa = (const float*)d_in[4];
    const float* wqc = (const float*)d_in[5];
    const float* bqc = (const float*)d_in[6];
    const float* wka = (const float*)d_in[7];
    const float* bka = (const float*)d_in[8];
    const float* wva = (const float*)d_in[9];
    const float* bva = (const float*)d_in[10];
    const float* wkc = (const float*)d_in[11];
    const float* bkc = (const float*)d_in[12];
    const float* wvc = (const float*)d_in[13];
    const float* bvc = (const float*)d_in[14];
    const float* wa  = (const float*)d_in[15];
    const float* ba  = (const float*)d_in[16];
    const float* wc  = (const float*)d_in[17];
    const float* bc  = (const float*)d_in[18];
    float* out = (float*)d_out;

    // workspace carve (all offsets 256B-aligned)
    char* ws = (char*)d_ws;
    float* pool = (float*)ws;                               // 16*2*64*84*4 = 688128
    size_t off = (size_t)BATCH * 2 * C_DIM * KREAL * 4;
    float* dvec = (float*)(ws + off);                       // 32*96*4 = 12288
    off += (size_t)BATCH * 2 * KP * 4;
    __bf16* Kqt = (__bf16*)(ws + off);                      // 32*96*64*2 = 393216
    off += (size_t)BATCH * 2 * KP * C_DIM * 2;
    __bf16* Mt = (__bf16*)(ws + off);                       // 32*64*96*2 = 393216

    psp_pool_kernel<<<BATCH * 2 * C_DIM, 256, 0, stream>>>(a, c, pool);

    prep_kernel<<<BATCH, 256, 0, stream>>>(0, pool, wqa, bqa, wka, bka, wva, bva,
                                           wa, ba, Kqt, dvec, Mt);
    prep_kernel<<<BATCH, 256, 0, stream>>>(1, pool, wqc, bqc, wkc, bkc, wvc, bvc,
                                           wc, bc, Kqt, dvec, Mt);

    const int tilesPerB = (HW + TILE - 1) / TILE;           // 345
    attn_kernel<<<BATCH * tilesPerB, 256, 0, stream>>>(x, a, c, Kqt, dvec, Mt, out);
}
